// QuantizedOldTransformer_30657476559606
// MI455X (gfx1250) — matrix operand order repaired and checked by reading, not compiled, not run on hardware
//
#include <hip/hip_runtime.h>

// ---------------------------------------------------------------------------
// CDNA5 (gfx1250) transformer forward. All GEMMs run on v_wmma_f32_16x16x32_bf16
// (wave32, 16x16x32 tiles, f32 accumulate). Model weights (~86MB) are L2-resident
// (192MB), so the chip rewards: every matmul on the WMMA path, vectorized
// b128 staging through LDS in fragment-native layout, fused epilogues.
// All GEMM launches here have M,N % 64 == 0 and K % 32 == 0 -> no bounds checks.
// transB is a template parameter so each variant is a straight-line K-loop.
// ---------------------------------------------------------------------------

typedef __attribute__((ext_vector_type(16))) __bf16 v16bf;
typedef __attribute__((ext_vector_type(8)))  __bf16 v8bf;
typedef __attribute__((ext_vector_type(8)))  float  v8f;
typedef __attribute__((ext_vector_type(4)))  float  v4f;

#define TB    256
#define BM    64
#define BN    64
#define BKK   32
#define LDP   40      // LDS row stride in bf16 (80B: 16B-aligned chunks, bank-skewed)

// Problem dims (from reference setup_inputs)
#define B_    2
#define S_    512
#define D_    1024
#define NL_   4
#define DFF_  2048
#define OUT_  1024
#define H_    16
#define DK_   64
#define VQN_  768
#define VQD_  64
#define SPW_  (S_ / 64)                 // flat-row segments per (b,d)
#define NE_   ((long)B_ * S_ * D_)      // 1,048,576 activation elements
#define BHSS_ ((long)B_ * H_ * S_ * S_) // 8,388,608 attention scores
#define NROW_ (B_ * D_ * SPW_)          // 16,384 VQ flat rows

// ---------------------------------------------------------------------------
// Generic batched WMMA GEMM: C = alpha*(A @ B(^T)) + bias + residual, opt ReLU.
// Block tile 64x64, 8 waves; wave (wm,wn) owns rows wm*16..+15, cols wn*32..+31
// => 1 A-fragment, 2 B-fragments, 2 WMMAs per 32-wide K step.
// LDS holds A as [m][k] and B transposed as [n][k] (k-contiguous) so each
// lane's WMMA fragment is contiguous 16B chunks -> ds_load_b128.
// Batch z offsets: off = (z/hdiv)*s1 + (z%hdiv)*s2  (hdiv=1 -> plain z*s1).
// Fragment layouts per CDNA5 ISA 7.12.2 (16-bit A 16x32, B 32x16, f32 C 16x16).
// ---------------------------------------------------------------------------
template <int TRANSB>
__global__ __launch_bounds__(TB)
void wmma_gemm_bf16(const float* __restrict__ A, const float* __restrict__ Bm,
                    const float* __restrict__ bias, const float* __restrict__ Res,
                    float* __restrict__ C,
                    int M, int N, int K, int lda, int ldb, int ldc,
                    long sA1, long sA2, long sB1, long sB2, long sC1, long sC2,
                    int hdiv, int relu, float alpha)
{
    __shared__ __bf16 As[BM][LDP];     // [m][k], k 0..31 used
    __shared__ __bf16 BsT[BN][LDP];    // [n][k], k 0..31 used

    const int z = blockIdx.z;
    const long offA = (long)(z / hdiv) * sA1 + (long)(z % hdiv) * sA2;
    const long offB = (long)(z / hdiv) * sB1 + (long)(z % hdiv) * sB2;
    const long offC = (long)(z / hdiv) * sC1 + (long)(z % hdiv) * sC2;
    A += offA; Bm += offB; C += offC;
    if (Res) Res += offC;

    const int row0 = blockIdx.y * BM;
    const int col0 = blockIdx.x * BN;
    const int tid  = threadIdx.x;
    const int wave = tid >> 5;           // 8 wave32 waves
    const int lane = tid & 31;
    const int wm   = wave & 3;           // 16-row group
    const int wn   = wave >> 2;          // 32-col group
    const int half = lane >> 4;
    const int l16  = lane & 15;

    // Staging assignment: thread t owns row (t>>2) and 8-wide k chunk ((t&3)*8)
    const int st_r  = tid >> 2;          // 0..63
    const int st_kc = (tid & 3) * 8;     // 0,8,16,24

    const int am  = wm * 16 + l16;       // A fragment row
    const int bn0 = wn * 32 + l16;       // first B fragment column

    // Hoisted base pointers; advanced by constant strides each K step.
    const float* ap = A + (long)(row0 + st_r) * lda + st_kc;
    const float* bp = TRANSB ? (Bm + (long)(col0 + st_r) * ldb + st_kc)
                             : (Bm + (long)st_kc * ldb + (col0 + st_r));
    const long bstep = TRANSB ? (long)BKK : (long)BKK * ldb;

    v8f acc0 = {};
    v8f acc1 = {};

    for (int k0 = 0; k0 < K; k0 += BKK) {
        // ---- Stage A tile: two float4 loads -> 8 bf16 -> one ds_store_b128
        {
            v4f f0 = *(const v4f*)ap;
            v4f f1 = *(const v4f*)(ap + 4);
            v8bf pk;
#pragma unroll
            for (int i = 0; i < 4; ++i) { pk[i] = (__bf16)f0[i]; pk[i + 4] = (__bf16)f1[i]; }
            *(v8bf*)&As[st_r][st_kc] = pk;
            ap += BKK;
        }
        // ---- Stage B tile transposed into [n][k]
        {
            v8bf pk;
            if (TRANSB) {   // B[n][k] in memory: contiguous along k
                v4f f0 = *(const v4f*)bp;
                v4f f1 = *(const v4f*)(bp + 4);
#pragma unroll
                for (int i = 0; i < 4; ++i) { pk[i] = (__bf16)f0[i]; pk[i + 4] = (__bf16)f1[i]; }
            } else {        // B[k][n] in memory: gather 8 k's (independent b32 loads)
#pragma unroll
                for (int i = 0; i < 8; ++i) pk[i] = (__bf16)bp[(long)i * ldb];
            }
            *(v8bf*)&BsT[st_r][st_kc] = pk;
            bp += bstep;
        }
        __syncthreads();

        // ---- Fragments (all contiguous 16B LDS chunks -> ds_load_b128)
        // A lane (half h): K h*16 .. h*16+15, two per VGPR
        v8bf a0 = *(const v8bf*)&As[am][half * 16];
        v8bf a1 = *(const v8bf*)&As[am][half * 16 + 8];
        v16bf af = __builtin_shufflevector(a0, a1, 0, 1, 2, 3, 4, 5, 6, 7,
                                           8, 9, 10, 11, 12, 13, 14, 15);
        // B lane (half h): K h*16 .. h*16+15, two per VGPR
        v8bf b00 = *(const v8bf*)&BsT[bn0][half * 16];
        v8bf b01 = *(const v8bf*)&BsT[bn0][half * 16 + 8];
        v16bf bf0 = __builtin_shufflevector(b00, b01, 0, 1, 2, 3, 4, 5, 6, 7,
                                            8, 9, 10, 11, 12, 13, 14, 15);
        v8bf b10 = *(const v8bf*)&BsT[bn0 + 16][half * 16];
        v8bf b11 = *(const v8bf*)&BsT[bn0 + 16][half * 16 + 8];
        v16bf bf1 = __builtin_shufflevector(b10, b11, 0, 1, 2, 3, 4, 5, 6, 7,
                                            8, 9, 10, 11, 12, 13, 14, 15);

        acc0 = __builtin_amdgcn_wmma_f32_16x16x32_bf16(false, af, false, bf0,
                                                       (short)0, acc0, false, false);
        acc1 = __builtin_amdgcn_wmma_f32_16x16x32_bf16(false, af, false, bf1,
                                                       (short)0, acc1, false, false);
        __syncthreads();
    }

    // Epilogue: C VGPR r holds (M = r + 8*half, N = l16) within each 16x16 tile
#pragma unroll
    for (int r = 0; r < 8; ++r) {
        int m  = row0 + wm * 16 + r + half * 8;
        int n0 = col0 + wn * 32 + l16;
        int n1 = n0 + 16;
        float v0 = alpha * acc0[r];
        float v1 = alpha * acc1[r];
        if (bias) { v0 += bias[n0]; v1 += bias[n1]; }
        if (Res)  { v0 += Res[(long)m * ldc + n0]; v1 += Res[(long)m * ldc + n1]; }
        if (relu) { v0 = fmaxf(v0, 0.0f); v1 = fmaxf(v1, 0.0f); }
        C[(long)m * ldc + n0] = v0;
        C[(long)m * ldc + n1] = v1;
    }
}

// ---------------------------------------------------------------------------
// LayerNorm (torch-style: unbiased std, eps added to std): one block per row.
// ---------------------------------------------------------------------------
__global__ __launch_bounds__(TB)
void layernorm_kernel(const float* __restrict__ x, const float* __restrict__ alpha,
                      const float* __restrict__ beta, float* __restrict__ y, int D)
{
    __shared__ float red[TB];
    const long row = blockIdx.x;
    const float* xr = x + row * D;
    const int tid = threadIdx.x;

    float s = 0.0f;
    for (int i = tid; i < D; i += TB) s += xr[i];
    red[tid] = s; __syncthreads();
    for (int o = TB / 2; o > 0; o >>= 1) { if (tid < o) red[tid] += red[tid + o]; __syncthreads(); }
    const float mean = red[0] / (float)D;
    __syncthreads();

    float v = 0.0f;
    for (int i = tid; i < D; i += TB) { float d = xr[i] - mean; v += d * d; }
    red[tid] = v; __syncthreads();
    for (int o = TB / 2; o > 0; o >>= 1) { if (tid < o) red[tid] += red[tid + o]; __syncthreads(); }
    const float var = red[0] / (float)(D - 1);
    const float inv = 1.0f / (sqrtf(var) + 1e-6f);

    float* yr = y + row * D;
    for (int i = tid; i < D; i += TB)
        yr[i] = alpha[i] * (xr[i] - mean) * inv + beta[i];
}

// ---------------------------------------------------------------------------
// Row softmax with pre-scale (1/sqrt(dk)), in place: one block per row.
// ---------------------------------------------------------------------------
__global__ __launch_bounds__(TB)
void softmax_kernel(float* __restrict__ x, int len, float scale)
{
    __shared__ float red[TB];
    float* xr = x + (long)blockIdx.x * len;
    const int tid = threadIdx.x;

    float m = -3.402823e38f;
    for (int i = tid; i < len; i += TB) m = fmaxf(m, xr[i] * scale);
    red[tid] = m; __syncthreads();
    for (int o = TB / 2; o > 0; o >>= 1) { if (tid < o) red[tid] = fmaxf(red[tid], red[tid + o]); __syncthreads(); }
    const float mx = red[0];
    __syncthreads();

    float s = 0.0f;
    for (int i = tid; i < len; i += TB) { float e = expf(xr[i] * scale - mx); xr[i] = e; s += e; }
    red[tid] = s; __syncthreads();
    for (int o = TB / 2; o > 0; o >>= 1) { if (tid < o) red[tid] += red[tid + o]; __syncthreads(); }
    const float inv = 1.0f / red[0];
    for (int i = tid; i < len; i += TB) xr[i] *= inv;
}

// ---------------------------------------------------------------------------
// VQ argmin: flat row r <-> (b, d, seg) of e[b, seg*64 : seg*64+64, d].
// One block per row; 768 codes split 3 per thread.
// ---------------------------------------------------------------------------
__global__ __launch_bounds__(TB)
void vq_argmin_kernel(const float* __restrict__ e, const float* __restrict__ emb,
                      int* __restrict__ idx)
{
    __shared__ float vec[VQD_];
    __shared__ float bd[TB];
    __shared__ int   bi[TB];
    const int row = blockIdx.x;
    const int seg = row % SPW_;
    const int d   = (row / SPW_) % D_;
    const int b   = row / (SPW_ * D_);
    const int tid = threadIdx.x;

    if (tid < VQD_)
        vec[tid] = e[(long)b * S_ * D_ + (long)(seg * 64 + tid) * D_ + d];
    __syncthreads();

    float best = 3.402823e38f; int bidx = 0;
    for (int c = tid; c < VQN_; c += TB) {
        const float* ec = emb + (long)c * VQD_;
        float dist = 0.0f;
#pragma unroll 8
        for (int i = 0; i < VQD_; ++i) { float t = vec[i] - ec[i]; dist += t * t; }
        if (dist < best) { best = dist; bidx = c; }
    }
    bd[tid] = best; bi[tid] = bidx; __syncthreads();
    for (int o = TB / 2; o > 0; o >>= 1) {
        if (tid < o && bd[tid + o] < bd[tid]) { bd[tid] = bd[tid + o]; bi[tid] = bi[tid + o]; }
        __syncthreads();
    }
    if (tid == 0) idx[row] = bi[0];
}

// ---------------------------------------------------------------------------
// VQ gather (straight-through == quantized values in fwd) + commit-loss sum.
// ---------------------------------------------------------------------------
__global__ __launch_bounds__(TB)
void vq_output_kernel(const float* __restrict__ e, const float* __restrict__ emb,
                      const int* __restrict__ idx, float* __restrict__ vq,
                      float* __restrict__ loss)
{
    __shared__ float red[TB];
    const long n = (long)blockIdx.x * TB + threadIdx.x;
    float d2 = 0.0f;
    if (n < NE_) {
        int d = (int)(n % D_);
        long bs = n / D_;
        int s = (int)(bs % S_);
        int b = (int)(bs / S_);
        int row = ((b * D_) + d) * SPW_ + (s >> 6);
        float q = emb[(long)idx[row] * VQD_ + (s & 63)];
        vq[n] = q;
        float t = q - e[n];
        d2 = t * t;
    }
    red[threadIdx.x] = d2; __syncthreads();
    for (int o = TB / 2; o > 0; o >>= 1) { if (threadIdx.x < o) red[threadIdx.x] += red[threadIdx.x + o]; __syncthreads(); }
    if (threadIdx.x == 0) atomicAdd(loss, red[0]);
}

__global__ void zero_scalar_kernel(float* p) { p[0] = 0.0f; }
__global__ void loss_final_kernel(const float* acc, float* out, float scale) { out[0] = acc[0] * scale; }

// ---------------------------------------------------------------------------
// Host-side orchestration helpers
// ---------------------------------------------------------------------------
static void launch_gemm(hipStream_t st, const float* A, const float* Bm, const float* bias,
                        const float* res, float* C, int M, int N, int K,
                        int lda, int ldb, int ldc,
                        long sA1, long sA2, long sB1, long sB2, long sC1, long sC2,
                        int batches, int hdiv, int transB, int relu, float alpha)
{
    dim3 grid((N + BN - 1) / BN, (M + BM - 1) / BM, batches);
    if (transB)
        wmma_gemm_bf16<1><<<grid, TB, 0, st>>>(A, Bm, bias, res, C, M, N, K, lda, ldb, ldc,
                                               sA1, sA2, sB1, sB2, sC1, sC2, hdiv, relu, alpha);
    else
        wmma_gemm_bf16<0><<<grid, TB, 0, st>>>(A, Bm, bias, res, C, M, N, K, lda, ldb, ldc,
                                               sA1, sA2, sB1, sB2, sC1, sC2, hdiv, relu, alpha);
}

static void launch_ln(hipStream_t st, const float* x, const float* p, float* y)
{
    layernorm_kernel<<<B_ * S_, TB, 0, st>>>(x, p, p + D_, y, D_);
}

// Multi-head attention with residual: X += OutProj(Attn(q_in, kv_in))
static void mha(hipStream_t st, const float* q_in, const float* kv_in,
                const float* w, const float* bvec, float* X,
                float* Q, float* Kb, float* Vb, float* ATT, float* AO)
{
    const int M = B_ * S_;
    // Q/K/V projections (full-D GEMMs with bias)
    launch_gemm(st, q_in,  w + 0L * D_ * D_, bvec + 0 * D_, nullptr, Q,  M, D_, D_, D_, D_, D_, 0, 0, 0, 0, 0, 0, 1, 1, 0, 0, 1.0f);
    launch_gemm(st, kv_in, w + 1L * D_ * D_, bvec + 1 * D_, nullptr, Kb, M, D_, D_, D_, D_, D_, 0, 0, 0, 0, 0, 0, 1, 1, 0, 0, 1.0f);
    launch_gemm(st, kv_in, w + 2L * D_ * D_, bvec + 2 * D_, nullptr, Vb, M, D_, D_, D_, D_, D_, 0, 0, 0, 0, 0, 0, 1, 1, 0, 0, 1.0f);
    // scores[b,h] = Qh @ Kh^T  (batched over B*H via head-sliced offsets)
    launch_gemm(st, Q, Kb, nullptr, nullptr, ATT, S_, S_, DK_, D_, D_, S_,
                (long)S_ * D_, DK_, (long)S_ * D_, DK_, (long)H_ * S_ * S_, (long)S_ * S_,
                B_ * H_, H_, 1, 0, 1.0f);
    softmax_kernel<<<B_ * H_ * S_, TB, 0, st>>>(ATT, S_, 0.125f);   // 1/sqrt(64)
    // O[b,h] = att @ Vh, written into concat-head layout directly
    launch_gemm(st, ATT, Vb, nullptr, nullptr, AO, S_, DK_, S_, S_, D_, D_,
                (long)H_ * S_ * S_, (long)S_ * S_, (long)S_ * D_, DK_, (long)S_ * D_, DK_,
                B_ * H_, H_, 0, 0, 1.0f);
    // X = X + AO @ Wout + bout (fused residual)
    launch_gemm(st, AO, w + 3L * D_ * D_, bvec + 3 * D_, X, X, M, D_, D_, D_, D_, D_, 0, 0, 0, 0, 0, 0, 1, 1, 0, 0, 1.0f);
}

static void ffn(hipStream_t st, const float* X2, const float* w1, const float* b1,
                const float* w2, const float* b2, float* X, float* FH)
{
    launch_gemm(st, X2, w1, b1, nullptr, FH, B_ * S_, DFF_, D_, D_, DFF_, DFF_, 0, 0, 0, 0, 0, 0, 1, 1, 0, 1, 1.0f); // ReLU
    launch_gemm(st, FH, w2, b2, X, X, B_ * S_, D_, DFF_, DFF_, D_, D_, 0, 0, 0, 0, 0, 0, 1, 1, 0, 0, 1.0f);          // residual
}

extern "C" void kernel_launch(void* const* d_in, const int* in_sizes, int n_in,
                              void* d_out, int out_size, void* d_ws, size_t ws_size,
                              hipStream_t stream)
{
    (void)in_sizes; (void)n_in; (void)out_size; (void)ws_size;
    const float* src       = (const float*)d_in[0];
    const float* trg       = (const float*)d_in[1];
    const float* enc_ln    = (const float*)d_in[2];   // (N,2,2,D)
    const float* enc_aw    = (const float*)d_in[3];   // (N,4,D,D)
    const float* enc_ab    = (const float*)d_in[4];   // (N,4,D)
    const float* enc_w1    = (const float*)d_in[5];
    const float* enc_b1    = (const float*)d_in[6];
    const float* enc_w2    = (const float*)d_in[7];
    const float* enc_b2    = (const float*)d_in[8];
    const float* enc_fn    = (const float*)d_in[9];   // (2,D)
    const float* dec_ln    = (const float*)d_in[10];  // (N,3,2,D)
    const float* dec_a1w   = (const float*)d_in[11];
    const float* dec_a1b   = (const float*)d_in[12];
    const float* dec_a2w   = (const float*)d_in[13];
    const float* dec_a2b   = (const float*)d_in[14];
    const float* dec_w1    = (const float*)d_in[15];
    const float* dec_b1    = (const float*)d_in[16];
    const float* dec_w2    = (const float*)d_in[17];
    const float* dec_b2    = (const float*)d_in[18];
    const float* dec_fn    = (const float*)d_in[19];
    const float* out_w     = (const float*)d_in[20];
    const float* out_b     = (const float*)d_in[21];
    const float* vq_emb    = (const float*)d_in[22];

    // Outputs (concatenated): output, e_outputs, vq_output, commit_loss
    float* OUTP  = (float*)d_out;
    float* EOUT  = OUTP + NE_;
    float* VQO   = EOUT + NE_;
    float* LOSSO = VQO + NE_;

    // Workspace layout (floats)
    float* ws  = (float*)d_ws;
    float* X   = ws;            // (B*S, D)
    float* X2  = X  + NE_;
    float* Q   = X2 + NE_;
    float* Kb  = Q  + NE_;
    float* Vb  = Kb + NE_;
    float* AO  = Vb + NE_;
    float* ATT = AO + NE_;      // (B*H, S, S)
    float* FH  = ATT + BHSS_;   // (B*S, DFF)
    int*   IDX = (int*)(FH + (long)B_ * S_ * DFF_);
    float* LACC = (float*)(IDX + NROW_);

    // ----------------------- Encoder -----------------------
    hipMemcpyAsync(X, src, NE_ * sizeof(float), hipMemcpyDeviceToDevice, stream);
    for (int l = 0; l < NL_; ++l) {
        const float* aw = enc_aw + (long)l * 4 * D_ * D_;
        const float* ab = enc_ab + (long)l * 4 * D_;
        launch_ln(stream, X, enc_ln + ((long)(l * 2 + 0) * 2) * D_, X2);
        mha(stream, X2, X2, aw, ab, X, Q, Kb, Vb, ATT, AO);
        launch_ln(stream, X, enc_ln + ((long)(l * 2 + 1) * 2) * D_, X2);
        ffn(stream, X2, enc_w1 + (long)l * D_ * DFF_, enc_b1 + (long)l * DFF_,
            enc_w2 + (long)l * DFF_ * D_, enc_b2 + (long)l * D_, X, FH);
    }
    launch_ln(stream, X, enc_fn, EOUT);   // e_outputs -> d_out segment 1

    // ----------------------- VQ quantizer -----------------------
    zero_scalar_kernel<<<1, 1, 0, stream>>>(LACC);
    vq_argmin_kernel<<<NROW_, TB, 0, stream>>>(EOUT, vq_emb, IDX);
    vq_output_kernel<<<(int)((NE_ + TB - 1) / TB), TB, 0, stream>>>(EOUT, vq_emb, IDX, VQO, LACC);
    loss_final_kernel<<<1, 1, 0, stream>>>(LACC, LOSSO, 0.25f / (float)NE_);

    // ----------------------- Decoder -----------------------
    hipMemcpyAsync(X, trg, NE_ * sizeof(float), hipMemcpyDeviceToDevice, stream);
    for (int l = 0; l < NL_; ++l) {
        launch_ln(stream, X, dec_ln + ((long)(l * 3 + 0) * 2) * D_, X2);
        mha(stream, X2, X2, dec_a1w + (long)l * 4 * D_ * D_, dec_a1b + (long)l * 4 * D_,
            X, Q, Kb, Vb, ATT, AO);
        launch_ln(stream, X, dec_ln + ((long)(l * 3 + 1) * 2) * D_, X2);
        mha(stream, X2, VQO, dec_a2w + (long)l * 4 * D_ * D_, dec_a2b + (long)l * 4 * D_,
            X, Q, Kb, Vb, ATT, AO);   // cross-attn: K,V from vq_output
        launch_ln(stream, X, dec_ln + ((long)(l * 3 + 2) * 2) * D_, X2);
        ffn(stream, X2, dec_w1 + (long)l * D_ * DFF_, dec_b1 + (long)l * DFF_,
            dec_w2 + (long)l * DFF_ * D_, dec_b2 + (long)l * D_, X, FH);
    }
    launch_ln(stream, X, dec_fn, X2);
    // output = d_output @ out_w + out_b -> d_out segment 0
    launch_gemm(stream, X2, out_w, out_b, nullptr, OUTP, B_ * S_, OUT_, D_,
                D_, OUT_, OUT_, 0, 0, 0, 0, 0, 0, 1, 1, 0, 0, 1.0f);
}
